// PosAttNorm_36240934044378
// MI455X (gfx1250) — matrix operand order repaired and checked by reading, not compiled, not run on hardware
//
#include <hip/hip_runtime.h>
#include <hip/hip_bf16.h>

#define BN  4
#define CC  256
#define LL  4096
#define NN  16
#define EPSF 1e-7f

typedef __attribute__((ext_vector_type(16))) __bf16 v16bf;
typedef __attribute__((ext_vector_type(8)))  float  v8f;
typedef __attribute__((ext_vector_type(4)))  int    v4i_t;

#if defined(__has_builtin)
#if __has_builtin(__builtin_amdgcn_global_load_async_to_lds_b128)
#define HAS_ASYNC_LDS 1
#endif
#if __has_builtin(__builtin_amdgcn_s_wait_asynccnt)
#define HAS_WAIT_ASYNC 1
#endif
#if __has_builtin(__builtin_amdgcn_sched_barrier)
#define SCHED_BARRIER() __builtin_amdgcn_sched_barrier(0)
#endif
#endif
#ifndef SCHED_BARRIER
#define SCHED_BARRIER()
#endif

#define AS1 __attribute__((address_space(1)))
#define AS3 __attribute__((address_space(3)))

// ---------- helpers ----------
__device__ __forceinline__ unsigned short f2bf(float f) {
    unsigned u = __float_as_uint(f);
    unsigned r = (u + 0x7FFFu + ((u >> 16) & 1u)) >> 16;   // RNE
    return (unsigned short)r;
}
__device__ __forceinline__ float uhash(unsigned seed, unsigned b, unsigned i) {
    unsigned h = seed ^ (b * 0x9E3779B9u) ^ (i * 0x85EBCA6Bu);
    h ^= h >> 16; h *= 0x7FEB352Du; h ^= h >> 15; h *= 0x846CA68Bu; h ^= h >> 16;
    return ((float)(h >> 8) + 0.5f) * (1.0f / 16777216.0f);
}
// 16-byte global -> LDS copy; async path (ASYNCcnt-tracked) when available.
__device__ __forceinline__ void copy16_g2l(const unsigned short* g, unsigned short* l) {
#if defined(HAS_ASYNC_LDS)
    __builtin_amdgcn_global_load_async_to_lds_b128(
        (AS1 v4i_t*)g, (AS3 v4i_t*)l, 0, 0);
#else
    *(uint4*)l = *(const uint4*)g;
#endif
}
__device__ __forceinline__ void wait_g2l() {
#if defined(HAS_ASYNC_LDS)
#if defined(HAS_WAIT_ASYNC)
    __builtin_amdgcn_s_wait_asynccnt(0);
#else
    asm volatile("s_wait_asynccnt 0" ::: "memory");
#endif
#endif
}

// ---------- fp32 -> bf16 pack (off the GEMM critical path) ----------
__global__ __launch_bounds__(256) void k_tobf(const float* __restrict__ src,
                                              unsigned short* __restrict__ dst, long n) {
    long i = ((long)blockIdx.x * 256 + threadIdx.x) * 8;
    if (i + 8 <= n) {
        float4 a = *(const float4*)(src + i);
        float4 b = *(const float4*)(src + i + 4);
        union { unsigned short h[8]; uint4 q; } pk;
        pk.h[0] = f2bf(a.x); pk.h[1] = f2bf(a.y); pk.h[2] = f2bf(a.z); pk.h[3] = f2bf(a.w);
        pk.h[4] = f2bf(b.x); pk.h[5] = f2bf(b.y); pk.h[6] = f2bf(b.z); pk.h[7] = f2bf(b.w);
        *(uint4*)(dst + i) = pk.q;
    }
}

// ---------- bf16 WMMA GEMM:  out[b] = A[b](MxK) @ X[b](KxL) (+bias) ----------
// WM waves along M (1 or 4); each wave computes a 16x64 strip (4 accumulators).
// Double-buffered LDS: stage K-tile i+1 (async A copy + B transpose-scatter)
// while the WMMAs consume K-tile i.
template <int WM>
__global__ __launch_bounds__(256) void k_gemm_bf(
    const unsigned short* __restrict__ A, const unsigned short* __restrict__ X,
    const float* __restrict__ bias, float* __restrict__ out,
    int M, int K, int L, long aBatchStride)
{
    constexpr int WNW = 8 / WM;
    constexpr int TM  = WM * 16;
    constexpr int TN  = WNW * 64;
    __shared__ unsigned short ldsA[2][TM * 32];     // row-major  TM x 32
    __shared__ unsigned short ldsBT[2][TN * 32];    // transposed TN x 32

    const int b    = blockIdx.z;
    const int m0   = blockIdx.y * TM;
    const int n0   = blockIdx.x * TN;
    const int tid  = threadIdx.x;
    const int wave = tid >> 5;
    const int lane = tid & 31;
    const int wm   = wave % WM;
    const int wn   = wave / WM;

    const unsigned short* Ab = A + (long)b * aBatchStride;
    const unsigned short* Xb = X + (long)b * (long)K * L;

    auto stage = [&](int k0, int buf) {
        // A tile (TM x 32 bf16) via async global->LDS, 16B per lane
        for (int ch = tid; ch < TM * 4; ch += 256) {
            int r = ch >> 2, c8 = (ch & 3) * 8;
            copy16_g2l(Ab + (long)(m0 + r) * K + k0 + c8, &ldsA[buf][r * 32 + c8]);
        }
        // B tile transposed: chunk = (k row, 8 cols)
        for (int ch = tid; ch < 32 * (TN / 8); ch += 256) {
            int kk = ch / (TN / 8);
            int nb = (ch % (TN / 8)) * 8;
            uint4 q = *(const uint4*)(Xb + (long)(k0 + kk) * L + n0 + nb);
            const unsigned short* hp = (const unsigned short*)&q;
            #pragma unroll
            for (int j = 0; j < 8; ++j)
                ldsBT[buf][(nb + j) * 32 + kk] = hp[j];
        }
    };

    v8f acc[4] = {v8f{0}, v8f{0}, v8f{0}, v8f{0}};

    stage(0, 0);

    for (int k0 = 0; k0 < K; k0 += 32) {
        const int cur = (k0 >> 5) & 1;
        wait_g2l();            // this wave's async copies into buf[cur] done
        __syncthreads();       // everyone's staging of buf[cur] visible
        if (k0 + 32 < K) {
            stage(k0 + 32, 1 - cur);   // overlaps with the WMMAs below
            if (k0 + 64 < K)
                __builtin_prefetch(Xb + (long)(k0 + 64) * L + n0, 0, 1);
        }

        // A fragment: lane<16 -> row=lane, K {0..7,16..23}; lane>=16 -> row=lane-16, K {8..15,24..31}
        const int row = wm * 16 + (lane & 15);
        const int klo = (lane < 16) ? 0 : 8;
        union FR { v16bf v; uint4 q[2]; };
        FR ua;
        ua.q[0] = *(const uint4*)&ldsA[cur][row * 32 + klo];
        ua.q[1] = *(const uint4*)&ldsA[cur][row * 32 + klo + 16];

        // load all 4 B fragments first, then issue the 4 WMMAs
        FR ub[4];
        const int off = (lane < 16) ? 0 : 8;
        #pragma unroll
        for (int s = 0; s < 4; ++s) {
            const int colI = wn * 64 + s * 16 + (lane & 15);
            const uint4* pb = (const uint4*)&ldsBT[cur][colI * 32 + off];
            ub[s].q[0] = pb[0]; ub[s].q[1] = pb[2];
        }
        SCHED_BARRIER();       // keep the ds_loads batched ahead of the WMMA group
        #pragma unroll
        for (int s = 0; s < 4; ++s)
            acc[s] = __builtin_amdgcn_wmma_f32_16x16x32_bf16(
                false, ua.v, false, ub[s].v, (short)0, acc[s], false, false);
    }

    // epilogue: D VGPR r -> M = r + (lane<16?0:8), N = lane%16
    const int moff = (lane < 16) ? 0 : 8;
    #pragma unroll
    for (int s = 0; s < 4; ++s) {
        const int col = n0 + wn * 64 + s * 16 + (lane & 15);
        #pragma unroll
        for (int r = 0; r < 8; ++r) {
            const int mm = m0 + wm * 16 + r + moff;
            float v = acc[s][r] + (bias ? bias[mm] : 0.0f);
            out[((long)b * M + mm) * (long)L + col] = v;
        }
    }
}

// ---------- orth loss ----------
__global__ void k_orth(const float* __restrict__ kn, float* __restrict__ dst) {
    __shared__ float sym[256];
    __shared__ float nrm[16];
    __shared__ float red[256];
    int tid = threadIdx.x, i = tid >> 4, j = tid & 15;
    float d = 0.f;
    for (int c = 0; c < CC; ++c) d += kn[i * CC + c] * kn[j * CC + c];
    sym[tid] = d; __syncthreads();
    if (tid < 16) nrm[tid] = sqrtf(sym[tid * 17]);
    __syncthreads();
    float lv = sym[tid] / (nrm[i] * nrm[j] + EPSF) - (i == j ? 1.f : 0.f);
    red[tid] = lv * lv; __syncthreads();
    for (int s = 128; s > 0; s >>= 1) { if (tid < s) red[tid] += red[tid + s]; __syncthreads(); }
    if (tid == 0) dst[0] = 0.001f * logf(red[0] + 1.0f);
}

// ---------- gumbel top-k (without replacement) ----------
__global__ void k_topk(const float* __restrict__ w, int wStride, int len, int k,
                       int invert, unsigned seed, int* __restrict__ outIdx, int outStride) {
    __shared__ float sc[LL];
    __shared__ float rv[256];
    __shared__ int   ri[256];
    int b = blockIdx.x, tid = threadIdx.x;
    for (int i = tid; i < len; i += 256) {
        float wv = w[b * wStride + i];
        if (invert) wv = 1.0f - wv;
        float u = uhash(seed, b, i);
        float g = -logf(-logf(u));
        sc[i] = logf(wv + 1e-20f) + g;
    }
    __syncthreads();
    for (int j = 0; j < k; ++j) {
        float best = -3.4e38f; int bi = len;
        for (int i = tid; i < len; i += 256)
            if (sc[i] > best) { best = sc[i]; bi = i; }
        rv[tid] = best; ri[tid] = bi; __syncthreads();
        for (int s = 128; s > 0; s >>= 1) {
            if (tid < s) {
                bool take = (rv[tid + s] > rv[tid]) ||
                            (rv[tid + s] == rv[tid] && ri[tid + s] < ri[tid]);
                if (take) { rv[tid] = rv[tid + s]; ri[tid] = ri[tid + s]; }
            }
            __syncthreads();
        }
        if (tid == 0) { outIdx[b * outStride + j] = ri[0]; sc[ri[0]] = -3.4e38f; }
        __syncthreads();
    }
}

// ---------- gather columns + column L2 norms ----------
__global__ void k_gather_norm(const float* __restrict__ src, const int* __restrict__ idx,
                              float* __restrict__ outT, float* __restrict__ norm, int kcount) {
    __shared__ float red[256];
    int b = blockIdx.x, tid = threadIdx.x;
    for (int j = 0; j < kcount; ++j) {
        int p = idx[b * 32 + j];
        float v = src[((long)(b * CC + tid)) * LL + p];
        outT[((long)b * kcount + j) * CC + tid] = v;
        red[tid] = v * v; __syncthreads();
        for (int s = 128; s > 0; s >>= 1) { if (tid < s) red[tid] += red[tid + s]; __syncthreads(); }
        if (tid == 0) norm[b * kcount + j] = sqrtf(red[0]);
        __syncthreads();
    }
}

// ---------- relation (32x32), row-max, softmax -> factor2 ----------
__global__ void k_relation(const float* __restrict__ fpfT, const float* __restrict__ fpxT,
                           const float* __restrict__ nf, const float* __restrict__ nx,
                           float* __restrict__ factor2) {
    __shared__ float rel[1024];
    __shared__ float rmax[32];
    int b = blockIdx.x, tid = threadIdx.x;
    for (int p = tid; p < 1024; p += 256) {
        int m = p >> 5, kk = p & 31;
        float d = 0.f;
        const float* pa = &fpfT[((long)b * 32 + m) * CC];
        const float* pb = &fpxT[((long)b * 32 + kk) * CC];
        for (int c = 0; c < CC; ++c) d += pa[c] * pb[c];
        rel[p] = d / (nf[b * 32 + m] * nx[b * 32 + kk] + EPSF);
    }
    __syncthreads();
    if (tid < 32) {
        float mx = -3.4e38f;
        for (int kk = 0; kk < 32; ++kk) mx = fmaxf(mx, rel[tid * 32 + kk]);
        rmax[tid] = mx;
    }
    __syncthreads();
    if (tid == 0) {
        float mx = -3.4e38f;
        for (int m = 0; m < 32; ++m) mx = fmaxf(mx, rmax[m]);
        float s = 0.f;
        for (int m = 0; m < 32; ++m) { float e = expf(rmax[m] - mx); rmax[m] = e; s += e; }
        float inv = 1.f / s;
        for (int m = 0; m < 32; ++m) factor2[b * 32 + m] = rmax[m] * inv;
    }
}

// ---------- select positions + gather filters_pos^T (bf16 for the act GEMM) ----------
__global__ void k_select(const float* __restrict__ fsa, const int* __restrict__ idxA,
                         const int* __restrict__ sel, int* __restrict__ spos,
                         unsigned short* __restrict__ fposTbf) {
    __shared__ int sp[16];
    int b = blockIdx.x, tid = threadIdx.x;
    if (tid < 16) { sp[tid] = idxA[b * 32 + sel[b * 16 + tid]]; spos[b * 16 + tid] = sp[tid]; }
    __syncthreads();
    for (int j = 0; j < 16; ++j)
        fposTbf[((long)b * 16 + j) * CC + tid] = f2bf(fsa[((long)(b * CC + tid)) * LL + sp[j]]);
}

// ---------- init ----------
__global__ void k_init(float* __restrict__ cnt) {
    if (threadIdx.x < BN * NN) cnt[threadIdx.x] = 0.f;
}

// ---------- attention softmax over N + per-n counts ----------
__global__ void k_att(const float* __restrict__ act, const float* __restrict__ xnb,
                      const float* __restrict__ alpha, float* __restrict__ att,
                      float* __restrict__ cnt) {
    __shared__ float ps[16];
    int tid = threadIdx.x, b = blockIdx.y;
    int l = blockIdx.x * 256 + tid;
    if (tid < 16) ps[tid] = 0.f;
    __syncthreads();
    float lg[16]; float mx = -3.4e38f;
    #pragma unroll
    for (int n = 0; n < 16; ++n) {
        float a = fminf(fmaxf(alpha[n], 0.f), 1.f);
        float v = (a * act[((long)b * NN + n) * LL + l] + xnb[((long)b * NN + n) * LL + l]) * 0.1f;
        lg[n] = v; mx = fmaxf(mx, v);
    }
    float s = 0.f;
    #pragma unroll
    for (int n = 0; n < 16; ++n) { float e = expf(lg[n] - mx); lg[n] = e; s += e; }
    float inv = 1.f / s;
    #pragma unroll
    for (int n = 0; n < 16; ++n) {
        float w = lg[n] * inv;
        att[((long)b * NN + n) * LL + l] = w;
        atomicAdd(&ps[n], w);
    }
    __syncthreads();
    if (tid < 16) atomicAdd(&cnt[b * NN + tid], ps[tid]);
}

// ---------- per-(b,c) weighted sums over L ----------
__global__ void k_stats(const float* __restrict__ xo, const float* __restrict__ att,
                        float* __restrict__ acc1, float* __restrict__ acc2) {
    __shared__ float r1[16], r2[16];
    int blk = blockIdx.x, b = blk >> 8, c = blk & 255;
    int tid = threadIdx.x, lane = tid & 31;
    float a1[16], a2[16];
    #pragma unroll
    for (int n = 0; n < 16; ++n) { a1[n] = 0.f; a2[n] = 0.f; }
    for (int l = tid; l < LL; l += 256) {
        float v = xo[((long)(b * CC + c)) * LL + l];
        float v2 = v * v;
        #pragma unroll
        for (int n = 0; n < 16; ++n) {
            float w = att[((long)b * NN + n) * LL + l];
            a1[n] += v * w; a2[n] += v2 * w * w;
        }
    }
    #pragma unroll
    for (int n = 0; n < 16; ++n)
        for (int off = 16; off > 0; off >>= 1) {
            a1[n] += __shfl_down(a1[n], off, 32);
            a2[n] += __shfl_down(a2[n], off, 32);
        }
    if (tid < 16) { r1[tid] = 0.f; r2[tid] = 0.f; }
    __syncthreads();
    if (lane == 0)
        #pragma unroll
        for (int n = 0; n < 16; ++n) { atomicAdd(&r1[n], a1[n]); atomicAdd(&r2[n], a2[n]); }
    __syncthreads();
    if (tid < 16) {
        acc1[((long)(b * CC + c)) * NN + tid] = r1[tid];
        acc2[((long)(b * CC + c)) * NN + tid] = r2[tid];
    }
}

// ---------- mean/std -> Ainv, Bm ----------
__global__ void k_finalize(const float* __restrict__ acc1, const float* __restrict__ acc2,
                           const float* __restrict__ cnt, float* __restrict__ Ainv,
                           float* __restrict__ Bm) {
    int i = blockIdx.x * 256 + threadIdx.x;               // over B*C*N = 16384
    int n = i & 15, b = i >> 12;
    float cn = cnt[b * NN + n] + EPSF;
    float s1 = acc1[i], s2 = acc2[i];
    float mean = s1 / cn;
    float vs = s2 - 2.f * mean * s1 + (float)LL * mean * mean;
    vs = fmaxf(vs, 0.f);
    float stdv = sqrtf(vs / cn);
    float ai = 1.f / (stdv + EPSF);
    Ainv[i] = ai;
    Bm[i]   = mean * ai;
}

// ---------- output: x + sigma * (xo*(Ainv@att) - Bm@att) ----------
__global__ __launch_bounds__(256) void k_out(
    const float* __restrict__ x, const float* __restrict__ xo,
    const float* __restrict__ att, const float* __restrict__ Ainv,
    const float* __restrict__ Bm, const float* __restrict__ sigma,
    float* __restrict__ out) {
    __shared__ float attL[16 * 256];
    __shared__ float aiS[CC * 16];
    __shared__ float bmS[CC * 16];
    int tid = threadIdx.x, b = blockIdx.y, l0 = blockIdx.x * 256;
    for (int p = tid; p < 4096; p += 256) {
        int n = p >> 8, ll = p & 255;
        attL[p] = att[((long)b * NN + n) * LL + l0 + ll];
        aiS[p]  = Ainv[(long)b * CC * NN + p];
        bmS[p]  = Bm[(long)b * CC * NN + p];
    }
    __syncthreads();
    float sg = sigma[0];
    int l = l0 + tid;
    for (int c = 0; c < CC; ++c) {
        float t1 = 0.f, t2 = 0.f;
        #pragma unroll
        for (int n = 0; n < 16; ++n) {
            float w = attL[n * 256 + tid];
            t1 += aiS[c * 16 + n] * w;
            t2 += bmS[c * 16 + n] * w;
        }
        long o = ((long)(b * CC + c)) * LL + l;
        out[o] = x[o] + sg * (xo[o] * t1 - t2);
    }
}

// =====================================================================
extern "C" void kernel_launch(void* const* d_in, const int* in_sizes, int n_in,
                              void* d_out, int out_size, void* d_ws, size_t ws_size,
                              hipStream_t stream) {
    const float* x     = (const float*)d_in[0];
    const float* f     = (const float*)d_in[1];
    const float* mask  = (const float*)d_in[2];
    const float* ksa_w = (const float*)d_in[3];
    const float* ksa_b = (const float*)d_in[4];
    const float* kr_w  = (const float*)d_in[5];
    const float* kr_b  = (const float*)d_in[6];
    const float* kn    = (const float*)d_in[7];
    const float* ko_w  = (const float*)d_in[8];
    const float* ko_b  = (const float*)d_in[9];
    const float* alpha = (const float*)d_in[10];
    const float* sigma = (const float*)d_in[11];
    float* out = (float*)d_out;

    // ---- workspace carve-up ----
    const size_t BCL = (size_t)BN * CC * LL;   // 4,194,304
    const size_t BNL = (size_t)BN * NN * LL;   //   262,144
    float* ws = (float*)d_ws;
    float* fsa   = ws;            ws += BCL;
    float* xr    = ws;            ws += BCL;
    float* xo    = ws;            ws += BCL;
    float* xnb   = ws;            ws += BNL;
    float* act   = ws;            ws += BNL;
    float* att   = ws;            ws += BNL;
    float* fpfT  = ws;            ws += (size_t)BN * 32 * CC;
    float* fpxT  = ws;            ws += (size_t)BN * 32 * CC;
    float* normf = ws;            ws += BN * 32;
    float* normx = ws;            ws += BN * 32;
    float* fac2  = ws;            ws += BN * 32;
    float* cnt   = ws;            ws += BN * NN + 16;      // pad to keep 16B alignment
    float* acc1  = ws;            ws += (size_t)BN * CC * NN;
    float* acc2  = ws;            ws += (size_t)BN * CC * NN;
    float* Ainv  = ws;            ws += (size_t)BN * CC * NN;
    float* Bm    = ws;            ws += (size_t)BN * CC * NN;
    unsigned short* xbf   = (unsigned short*)ws;
    unsigned short* fbf   = xbf + BCL;
    unsigned short* xrbf  = fbf + BCL;
    unsigned short* wksa  = xrbf + BCL;
    unsigned short* wkr   = wksa + CC * CC;
    unsigned short* wko   = wkr + CC * CC;
    unsigned short* knbf  = wko + CC * CC;
    unsigned short* fposTbf = knbf + NN * CC;
    int* idxA = (int*)(fposTbf + (size_t)BN * NN * CC);
    int* idxX = idxA + BN * 32;
    int* sel  = idxX + BN * 32;
    int* spos = sel + BN * 16;

    dim3 blk(256);

    // ---- bf16 pre-conversion (bandwidth-trivial) ----
    k_tobf<<<BCL / 2048, blk, 0, stream>>>(x, xbf, (long)BCL);
    k_tobf<<<BCL / 2048, blk, 0, stream>>>(f, fbf, (long)BCL);
    k_tobf<<<(CC * CC) / 2048, blk, 0, stream>>>(ksa_w, wksa, CC * CC);
    k_tobf<<<(CC * CC) / 2048, blk, 0, stream>>>(kr_w,  wkr,  CC * CC);
    k_tobf<<<(CC * CC) / 2048, blk, 0, stream>>>(ko_w,  wko,  CC * CC);
    k_tobf<<<(NN * CC) / 2048 + 1, blk, 0, stream>>>(kn, knbf, NN * CC);

    // ---- big WMMA GEMMs: macro tile 64x128, grid (L/128, M/64, B) ----
    k_gemm_bf<4><<<dim3(LL / 128, CC / 64, BN), blk, 0, stream>>>(wksa, fbf, ksa_b, fsa, CC, CC, LL, 0);
    k_gemm_bf<4><<<dim3(LL / 128, CC / 64, BN), blk, 0, stream>>>(wkr,  xbf, kr_b,  xr,  CC, CC, LL, 0);
    k_gemm_bf<4><<<dim3(LL / 128, CC / 64, BN), blk, 0, stream>>>(wko,  xbf, ko_b,  xo,  CC, CC, LL, 0);
    // skinny M=16 GEMM: macro tile 16x512, grid (L/512, 1, B)
    k_gemm_bf<1><<<dim3(LL / 512, 1, BN), blk, 0, stream>>>(knbf, xbf, nullptr, xnb, NN, CC, LL, 0);

    k_orth<<<1, blk, 0, stream>>>(kn, out + BCL);
    k_init<<<1, blk, 0, stream>>>(cnt);

    k_topk<<<BN, blk, 0, stream>>>(mask, LL, LL, 32, 0, 0x243F6A88u, idxA, 32);
    k_topk<<<BN, blk, 0, stream>>>(mask, LL, LL, 32, 1, 0x85A308D3u, idxX, 32);
    k_gather_norm<<<BN, blk, 0, stream>>>(fsa, idxA, fpfT, normf, 32);
    k_gather_norm<<<BN, blk, 0, stream>>>(xr,  idxX, fpxT, normx, 32);
    k_relation<<<BN, blk, 0, stream>>>(fpfT, fpxT, normf, normx, fac2);
    k_topk<<<BN, blk, 0, stream>>>(fac2, 32, 32, 16, 0, 0x13198A2Eu, sel, 16);
    k_select<<<BN, blk, 0, stream>>>(fsa, idxA, sel, spos, fposTbf);

    // act = filters_pos^T @ xr  (per-batch bf16 A, stride 16*256)
    k_tobf<<<BCL / 2048, blk, 0, stream>>>(xr, xrbf, (long)BCL);
    k_gemm_bf<1><<<dim3(LL / 512, 1, BN), blk, 0, stream>>>(fposTbf, xrbf, nullptr, act, NN, CC, LL, (long)NN * CC);

    k_att<<<dim3(LL / 256, BN), blk, 0, stream>>>(act, xnb, alpha, att, cnt);
    k_stats<<<BN * CC, blk, 0, stream>>>(xo, att, acc1, acc2);
    k_finalize<<<(BN * CC * NN) / 256, blk, 0, stream>>>(acc1, acc2, cnt, Ainv, Bm);
    k_out<<<dim3(LL / 256, BN), blk, 0, stream>>>(x, xo, att, Ainv, Bm, sigma, out);
}
